// MultiHeadAdaIn_2293512536915
// MI455X (gfx1250) — matrix operand order repaired and checked by reading, not compiled, not run on hardware
//
#include <hip/hip_runtime.h>
#include <hip/hip_bf16.h>

#define Bn 4
#define Nn 8192
#define MD 512
#define Hh 8
#define Ff 16
#define Ss 32
#define S3 32768
#define Ll 256
#define CKV 152          // H*(F+3)
#define CKVP 160         // padded to 10 WMMA m-tiles
#define CHF 128          // H*F
#define EPSF 1e-5f

typedef __attribute__((ext_vector_type(16))) __bf16 v16bf;
typedef __attribute__((ext_vector_type(8)))  float  v8f;
typedef __attribute__((ext_vector_type(8)))  unsigned v8u;
typedef __attribute__((ext_vector_type(4)))  int    v4i;
typedef __attribute__((ext_vector_type(4)))  float  v4f;

static __device__ __forceinline__ __bf16 f2bf(float f) {
    union { float f; unsigned u; } v; v.f = f;
    unsigned r = (v.u + 0x7FFFu + ((v.u >> 16) & 1u)) >> 16;
    union { unsigned short s; __bf16 b; } o; o.s = (unsigned short)r;
    return o.b;
}

// compile-time LDS offset for conv patch element k (folds under full unroll)
// LDS tile: [fi(16)][d0i(3)][c1i(6)][c2p(34)]; c2 padded with zero edge slots
static __device__ __forceinline__ constexpr int ldsOff(int k) {
    int fi = k / 27, tap = k % 27;
    int d0i = tap / 9, r = tap % 9;
    int d1i = r / 3, d2 = r % 3 - 1;
    return fi * 612 + d0i * 204 + d1i * 34 + d2 + 1;
}

// ---------------------------------------------------------------------------
// Kernel 0a: convert input activations to bf16 once
// ---------------------------------------------------------------------------
__global__ void k_cvtX(const float* __restrict__ X, __bf16* __restrict__ Xb) {
    size_t gid = (size_t)blockIdx.x * 256 + threadIdx.x;
    if (gid < (size_t)Bn * MD * Nn) Xb[gid] = f2bf(X[gid]);
}

// ---------------------------------------------------------------------------
// Kernel 0b: pack W_kv into WMMA-A fragment order, bf16, zero-padded to 160
// rows. Layout: [mt(10)][kc(16)][lane(32)][e(16)].
// ---------------------------------------------------------------------------
__global__ void k_packA(const float* __restrict__ W, __bf16* __restrict__ aW) {
    int gid = blockIdx.x * 256 + threadIdx.x;      // 81920 total
    if (gid >= 81920) return;
    int e = gid & 15, lane = (gid >> 4) & 31, kc = (gid >> 9) & 15, mt = gid >> 13;
    int m = lane & 15, kh = lane >> 4, j = e >> 1, t = e & 1;
    int k = kc * 32 + kh * 16 + e;
    int row = mt * 16 + m;
    float v = (row < CKV) ? W[row * MD + k] : 0.f;
    aW[gid] = f2bf(v);
}

// ---------------------------------------------------------------------------
// Kernel 0c: pack grouped-conv weights into WMMA-A fragment order, bf16,
// K padded 432 -> 448. Layout: [h(8)][kc(14)][lane(32)][e(16)].
// ---------------------------------------------------------------------------
__global__ void k_packWc(const float* __restrict__ cw, __bf16* __restrict__ wA) {
    int gid = blockIdx.x * 256 + threadIdx.x;      // 57344 total
    if (gid >= 57344) return;
    int e = gid & 15; int r = gid >> 4; int lane = r & 31; r >>= 5;
    int kc = r % 14; int h = r / 14;
    int m = lane & 15, kh = lane >> 4, j = e >> 1, t = e & 1;
    int k = kc * 32 + kh * 16 + e;
    float v = (k < 432) ? cw[(h * 16 + m) * 432 + k] : 0.f;
    wA[gid] = f2bf(v);
}

// ---------------------------------------------------------------------------
// Kernel 1: kv = W_kv x input via WMMA bf16 (fp32 accum).
// One wave = 5 m-tiles x 16 cols; B loaded once per K-chunk; prefetch next B.
// ---------------------------------------------------------------------------
__global__ __launch_bounds__(128) void k_gemm_kv(const __bf16* __restrict__ aW,
                                                 const __bf16* __restrict__ Xb,
                                                 float* __restrict__ kv) {
    const int NT = Nn / 16;
    int wave = (blockIdx.x * blockDim.x + threadIdx.x) >> 5;
    int lane = threadIdx.x & 31;
    int mg = wave & 1;                  // m-group: tiles mg*5 .. mg*5+4
    int nt = (wave >> 1) % NT;
    int b  = (wave >> 1) / NT;
    int m  = lane & 15;
    int kh = lane >> 4;
    const __bf16* xrow = Xb + (size_t)b * MD * Nn + nt * 16;
    const __bf16* xb   = xrow + m;
    v8f acc0 = {}, acc1 = {}, acc2 = {}, acc3 = {}, acc4 = {};
    for (int kc = 0; kc < MD / 32; ++kc) {
        v16bf bm;
#pragma unroll
        for (int j = 0; j < 8; ++j) {
            int kb = kc * 32 + kh * 16 + 2 * j;   // B: lanes 0-15 K lo, 16-31 K hi
            bm[2 * j]     = xb[(size_t)kb * Nn];
            bm[2 * j + 1] = xb[(size_t)(kb + 1) * Nn];
        }
        if (kc < 15)   // software prefetch next chunk's 32 B-rows (1 line each)
            __builtin_prefetch(xrow + (size_t)((kc + 1) * 32 + lane) * Nn, 0, 0);
        const v16bf* ap = (const v16bf*)aW + ((size_t)(mg * 80 + kc) * 32 + lane);
        v16bf a0 = ap[0 * 512];
        acc0 = __builtin_amdgcn_wmma_f32_16x16x32_bf16(false, a0, false, bm, (short)0, acc0, false, false);
        v16bf a1 = ap[1 * 512];
        acc1 = __builtin_amdgcn_wmma_f32_16x16x32_bf16(false, a1, false, bm, (short)0, acc1, false, false);
        v16bf a2 = ap[2 * 512];
        acc2 = __builtin_amdgcn_wmma_f32_16x16x32_bf16(false, a2, false, bm, (short)0, acc2, false, false);
        v16bf a3 = ap[3 * 512];
        acc3 = __builtin_amdgcn_wmma_f32_16x16x32_bf16(false, a3, false, bm, (short)0, acc3, false, false);
        v16bf a4 = ap[4 * 512];
        acc4 = __builtin_amdgcn_wmma_f32_16x16x32_bf16(false, a4, false, bm, (short)0, acc4, false, false);
    }
    int ncol = nt * 16 + m;
    float* kvb = kv + (size_t)b * CKVP * Nn + ncol;
#pragma unroll
    for (int v = 0; v < 8; ++v) {
        int r0 = mg * 80 + v + kh * 8;            // C/D layout row
        kvb[(size_t)(r0)      * Nn] = acc0[v];
        kvb[(size_t)(r0 + 16) * Nn] = acc1[v];
        kvb[(size_t)(r0 + 32) * Nn] = acc2[v];
        kvb[(size_t)(r0 + 48) * Nn] = acc3[v];
        kvb[(size_t)(r0 + 64) * Nn] = acc4[v];
    }
}

// ---------------------------------------------------------------------------
// Kernel 2: style projections  h = style @ w.T + b  (tiny: 2240 dot-256)
// ---------------------------------------------------------------------------
__global__ void k_style(const float* __restrict__ style,
                        const float* __restrict__ kw_w, const float* __restrict__ kw_b,
                        const float* __restrict__ vw_w, const float* __restrict__ vw_b,
                        const float* __restrict__ aw_w, const float* __restrict__ aw_b,
                        float* __restrict__ kgb, float* __restrict__ vgb,
                        float* __restrict__ agb) {
    const int per = 48 + 256 + 256;
    int gid = blockIdx.x * blockDim.x + threadIdx.x;
    if (gid >= Bn * per) return;
    int b = gid / per, r = gid % per;
    const float* w; const float* bias; float* out; int idx;
    if (r < 48)            { w = kw_w; bias = kw_b; out = kgb + b * 48;  idx = r; }
    else if (r < 48 + 256) { w = vw_w; bias = vw_b; out = vgb + b * 256; idx = r - 48; }
    else                   { w = aw_w; bias = aw_b; out = agb + b * 256; idx = r - 304; }
    const float* st = style + b * Ll;
    float s = bias[idx];
    for (int i = 0; i < Ll; ++i) s += st[i] * w[idx * Ll + i];
    out[idx] = s;
}

// ---------------------------------------------------------------------------
// Kernel 3: per-(b,channel) mean / rsqrt(var+eps) over N. x is [B][Cpad][N].
// ---------------------------------------------------------------------------
__global__ __launch_bounds__(256) void k_stats(const float* __restrict__ x,
                                               float* __restrict__ mu,
                                               float* __restrict__ rinv,
                                               int C, int Cpad) {
    int bc = blockIdx.x;
    int b = bc / C, ch = bc % C;
    const float* p = x + ((size_t)b * Cpad + ch) * Nn;
    float s = 0.f, ss = 0.f;
    for (int i = threadIdx.x; i < Nn; i += 256) { float v = p[i]; s += v; ss += v * v; }
    __shared__ float sh1[256], sh2[256];
    sh1[threadIdx.x] = s; sh2[threadIdx.x] = ss;
    __syncthreads();
    for (int o = 128; o > 0; o >>= 1) {
        if (threadIdx.x < o) {
            sh1[threadIdx.x] += sh1[threadIdx.x + o];
            sh2[threadIdx.x] += sh2[threadIdx.x + o];
        }
        __syncthreads();
    }
    if (threadIdx.x == 0) {
        float m = sh1[0] / Nn;
        float v = sh2[0] / Nn - m * m;
        mu[bc] = m;
        rinv[bc] = rsqrtf(v + EPSF);
    }
}

// ---------------------------------------------------------------------------
// Kernel 4: keys AdaIN -> points -> tanh lattice -> trilinear splat (atomics)
// ---------------------------------------------------------------------------
__global__ void k_splat(const float* __restrict__ kv, const float* __restrict__ orig,
                        const float* __restrict__ Tw, const float* __restrict__ Tb,
                        const float* __restrict__ scal,
                        const float* __restrict__ kgb, const float* __restrict__ vgb,
                        const float* __restrict__ mu, const float* __restrict__ rinv,
                        int* __restrict__ idxbuf, float* __restrict__ wbuf,
                        float* __restrict__ z) {
    int gid = blockIdx.x * blockDim.x + threadIdx.x;
    if (gid >= Bn * Hh * Nn) return;
    int n = gid % Nn, h = (gid / Nn) % Hh, b = gid / (Nn * Hh);
    float scale = scal[0];
    float pt[3];
#pragma unroll
    for (int c = 0; c < 3; ++c) {
        int ch = h * 3 + c;
        float x  = kv[((size_t)b * CKVP + ch) * Nn + n];
        float xn = (x - mu[b * CKV + ch]) * rinv[b * CKV + ch];
        float kr = (1.f + kgb[b * 48 + ch]) * xn + kgb[b * 48 + 24 + ch];
        pt[c] = orig[((size_t)b * 3 + c) * Nn + n] + scale * kr;
    }
    float f0[3], t[3];
#pragma unroll
    for (int i = 0; i < 3; ++i) {
        float key = Tw[h * 9 + i * 3 + 0] * pt[0] + Tw[h * 9 + i * 3 + 1] * pt[1]
                  + Tw[h * 9 + i * 3 + 2] * pt[2] + Tb[h * 3 + i];
        float g = (tanhf(key) + 1.f) * 0.5f * (float)(Ss - 1);
        float fl = fminf(fmaxf(floorf(g), 0.f), (float)(Ss - 2));
        f0[i] = fl;
        t[i]  = fminf(fmaxf(g - fl, 0.f), 1.f);
    }
    int bh = b * Hh + h;
    int idx8[8]; float w8[8];
#pragma unroll
    for (int c = 0; c < 8; ++c) {
        int a0 = (c >> 2) & 1, a1 = (c >> 1) & 1, a2 = c & 1;
        int i0 = (int)f0[0] + a0, i1 = (int)f0[1] + a1, i2 = (int)f0[2] + a2;
        float w = (a0 ? t[0] : 1.f - t[0]) * (a1 ? t[1] : 1.f - t[1])
                * (a2 ? t[2] : 1.f - t[2]);
        int flat = (i0 * Ss + i1) * Ss + i2;
        idx8[c] = flat; w8[c] = w;
        idxbuf[((size_t)bh * Nn + n) * 8 + c] = flat;
        wbuf[((size_t)bh * Nn + n) * 8 + c]  = w;
    }
    for (int f = 0; f < Ff; ++f) {
        int ch = 24 + h * 16 + f;
        float x   = kv[((size_t)b * CKVP + ch) * Nn + n];
        float xn  = (x - mu[b * CKV + ch]) * rinv[b * CKV + ch];
        int vc    = h * 16 + f;
        float val = (1.f + vgb[b * 256 + vc]) * xn + vgb[b * 256 + 128 + vc];
        float* zb = z + ((size_t)(bh * 16 + f) << 15);
#pragma unroll
        for (int c = 0; c < 8; ++c) atomicAdd(zb + idx8[c], w8[c] * val);
    }
}

// ---------------------------------------------------------------------------
// Kernel 6: grouped 3x3x3 conv, implicit GEMM via WMMA, ASYNC double-buffered
// LDS staging. Block = (bh, c0), persistent over 8 c1-quads: while 8 waves
// run the WMMAs for tile q, global_load_async_to_lds_b32 streams tile q+1
// into the other LDS buffer (ASYNCcnt); one s_wait_asynccnt + one barrier per
// tile. Fragments: fp32 pairs from LDS packed to bf16 via one v_perm_b32.
// ---------------------------------------------------------------------------
__global__ __launch_bounds__(256) void k_conv(const __bf16* __restrict__ wA,
                                              const float* __restrict__ z,
                                              const float* __restrict__ cb,
                                              float* __restrict__ zc) {
    __shared__ float stg[2][16 * 3 * 6 * 34];     // 2 x 9792 fp32 = 78 KB
    int bid = blockIdx.x;
    int c0 = bid & 31, bh = bid >> 5;
    int h = bh & 7;
    int tid = threadIdx.x;

    auto issue_tile = [&](int q, int buf) {
        int c1b = q * 4;
        for (int idx = tid; idx < 9792; idx += 256) {
            int c2p = idx % 34; int rem = idx / 34;
            int c1i = rem % 6;  rem /= 6;
            int d0i = rem % 3;  int fi = rem / 3;
            int g0 = c0 + d0i - 1, g1 = c1b + c1i - 1, g2 = c2p - 1;
            if ((unsigned)g0 < 32u && (unsigned)g1 < 32u && (unsigned)g2 < 32u) {
                const float* gp = z + ((size_t)(bh * 16 + fi) << 15)
                                    + (g0 * 32 + g1) * 32 + g2;
                // LDS aperture addresses carry the LDS byte offset in [31:0]
                unsigned ldsa = (unsigned)(size_t)&stg[buf][idx];
                asm volatile("global_load_async_to_lds_b32 %0, %1, off"
                             :: "v"(ldsa), "v"(gp) : "memory");
            } else {
                stg[buf][idx] = 0.f;               // zero halo slot (ds_store)
            }
        }
    };

    int w = tid >> 5, lane = tid & 31;
    int c2b = (w & 1) * 16;
    int m = lane & 15;
    bool khi = (lane >> 4) != 0;
    int base = (w >> 1) * 34 + c2b + m;

    issue_tile(0, 0);
    for (int q = 0; q < 8; ++q) {
        int buf = q & 1;
        asm volatile("s_wait_asynccnt 0x0" ::: "memory");
        __syncthreads();
        if (q < 7) issue_tile(q + 1, buf ^ 1);
        const float* sb = stg[buf];
        v8f acc = {};
#pragma unroll
        for (int kc = 0; kc < 13; ++kc) {          // k = kc*32 + e (+16) < 432
            v16bf a = ((const v16bf*)wA)[(h * 14 + kc) * 32 + lane];
            v8u bu;
#pragma unroll
            for (int j = 0; j < 8; ++j) {
                int k0 = kc * 32 + 2 * j;          // compile-time
                int off0 = khi ? ldsOff(k0 + 16) : ldsOff(k0);
                int off1 = khi ? ldsOff(k0 + 17) : ldsOff(k0 + 1);
                unsigned x0 = __builtin_bit_cast(unsigned, sb[base + off0]);
                unsigned x1 = __builtin_bit_cast(unsigned, sb[base + off1]);
                bu[j] = __builtin_amdgcn_perm(x1, x0, 0x07060302u); // hi16 pair
            }
            v16bf bm = __builtin_bit_cast(v16bf, bu);
            acc = __builtin_amdgcn_wmma_f32_16x16x32_bf16(false, a, false, bm,
                                                          (short)0, acc, false, false);
        }
        {   // peeled kc = 13: upper lane-half covers k 432..447 -> zeros
            v16bf a = ((const v16bf*)wA)[(h * 14 + 13) * 32 + lane];
            v8u bu;
#pragma unroll
            for (int j = 0; j < 8; ++j) {
                int k0 = 13 * 32 + 2 * j;          // 416..430
                unsigned x0 = __builtin_bit_cast(unsigned, sb[base + ldsOff(k0)]);
                unsigned x1 = __builtin_bit_cast(unsigned, sb[base + ldsOff(k0 + 1)]);
                unsigned p = __builtin_amdgcn_perm(x1, x0, 0x07060302u);
                bu[j] = khi ? 0u : p;
            }
            v16bf bm = __builtin_bit_cast(v16bf, bu);
            acc = __builtin_amdgcn_wmma_f32_16x16x32_bf16(false, a, false, bm,
                                                          (short)0, acc, false, false);
        }
        int c1 = q * 4 + (w >> 1);
        size_t pbase = (size_t)(c0 * 32 + c1) * 32 + c2b + m;
#pragma unroll
        for (int v = 0; v < 8; ++v) {
            int fo = v + (khi ? 8 : 0);
            zc[((size_t)(bh * 16 + fo) << 15) + pbase] = acc[v] + cb[h * 16 + fo];
        }
    }
}

// ---------------------------------------------------------------------------
// Kernel 7: weighted gather (slice) back to per-point features
// ---------------------------------------------------------------------------
__global__ void k_slice(const int* __restrict__ idxbuf, const float* __restrict__ wbuf,
                        const float* __restrict__ zc, float* __restrict__ sliced) {
    int gid = blockIdx.x * blockDim.x + threadIdx.x;
    if (gid >= Bn * Hh * Nn) return;
    int n = gid % Nn, bh = gid / Nn;
    const v4i* ip = (const v4i*)(idxbuf + ((size_t)bh * Nn + n) * 8);
    const v4f* wp = (const v4f*)(wbuf   + ((size_t)bh * Nn + n) * 8);
    v4i i0 = ip[0], i1 = ip[1];
    v4f w0 = wp[0], w1 = wp[1];
    for (int f = 0; f < Ff; ++f) {
        const float* zb = zc + ((size_t)(bh * 16 + f) << 15);
        float s = 0.f;
#pragma unroll
        for (int c = 0; c < 4; ++c) s += w0[c] * zb[i0[c]];
#pragma unroll
        for (int c = 0; c < 4; ++c) s += w1[c] * zb[i1[c]];
        sliced[((size_t)(bh * 16 + f)) * Nn + n] = s;  // == [b][h*16+f][n]
    }
}

// ---------------------------------------------------------------------------
// Kernel 8: final AdaIN + ReLU
// ---------------------------------------------------------------------------
__global__ void k_final(const float* __restrict__ sliced, const float* __restrict__ agb,
                        const float* __restrict__ mu, const float* __restrict__ rinv,
                        float* __restrict__ out) {
    size_t gid = (size_t)blockIdx.x * blockDim.x + threadIdx.x;
    if (gid >= (size_t)Bn * CHF * Nn) return;
    int ch = (int)((gid / Nn) % CHF);
    int b  = (int)(gid / ((size_t)Nn * CHF));
    float xn = (sliced[gid] - mu[b * CHF + ch]) * rinv[b * CHF + ch];
    float r  = (1.f + agb[b * 256 + ch]) * xn + agb[b * 256 + 128 + ch];
    out[gid] = fmaxf(r, 0.f);
}

// ---------------------------------------------------------------------------
extern "C" void kernel_launch(void* const* d_in, const int* in_sizes, int n_in,
                              void* d_out, int out_size, void* d_ws, size_t ws_size,
                              hipStream_t stream) {
    const float* input  = (const float*)d_in[0];
    const float* style  = (const float*)d_in[1];
    const float* orig   = (const float*)d_in[2];
    const float* W_kv   = (const float*)d_in[3];
    const float* kw_w   = (const float*)d_in[4];
    const float* kw_b   = (const float*)d_in[5];
    const float* vw_w   = (const float*)d_in[6];
    const float* vw_b   = (const float*)d_in[7];
    const float* aw_w   = (const float*)d_in[8];
    const float* aw_b   = (const float*)d_in[9];
    const float* scal   = (const float*)d_in[10];
    const float* T_w    = (const float*)d_in[11];
    const float* T_b    = (const float*)d_in[12];
    const float* conv_w = (const float*)d_in[13];
    const float* conv_b = (const float*)d_in[14];

    char* base = (char*)d_ws;
    size_t off = 0;
    auto take = [&](size_t bytes) -> char* {
        char* p = base + off;
        off = (off + bytes + 255) & ~(size_t)255;
        return p;
    };
    float*  kv     = (float*) take((size_t)Bn * CKVP * Nn * 4);      // ~21 MB
    float*  kgb    = (float*) take((size_t)Bn * 48 * 4);
    float*  vgb    = (float*) take((size_t)Bn * 256 * 4);
    float*  agb    = (float*) take((size_t)Bn * 256 * 4);
    float*  mu1    = (float*) take((size_t)Bn * CKV * 4);
    float*  rinv1  = (float*) take((size_t)Bn * CKV * 4);
    int*    idxbuf = (int*)   take((size_t)Bn * Hh * Nn * 8 * 4);    // 8 MB
    float*  wbuf   = (float*) take((size_t)Bn * Hh * Nn * 8 * 4);    // 8 MB
    float*  z      = (float*) take((size_t)Bn * Hh * Ff * S3 * 4);   // 64 MB
    float*  zc     = (float*) take((size_t)Bn * Hh * Ff * S3 * 4);   // 64 MB
    __bf16* Xb     = (__bf16*)take((size_t)Bn * MD * Nn * 2);        // 32 MB
    __bf16* aW     = (__bf16*)take((size_t)81920 * 2);
    __bf16* wA     = (__bf16*)take((size_t)57344 * 2);
    float*  sliced = (float*) take((size_t)Bn * CHF * Nn * 4);       // 16 MB
    float*  mu2    = (float*) take((size_t)Bn * CHF * 4);
    float*  rinv2  = (float*) take((size_t)Bn * CHF * 4);

    // 0) precision/layout prep: bf16 activations + fragment-packed weights
    k_cvtX <<<(int)(((size_t)Bn * MD * Nn) / 256), 256, 0, stream>>>(input, Xb);
    k_packA<<<81920 / 256, 256, 0, stream>>>(W_kv, aW);
    k_packWc<<<57344 / 256, 256, 0, stream>>>(conv_w, wA);

    // 1) kv projection GEMM (WMMA bf16): 4096 waves x 5 m-tiles each
    k_gemm_kv<<<(Bn * (Nn / 16) * 2) / 4, 128, 0, stream>>>(aW, Xb, kv);

    // 2) style projections
    k_style<<<(Bn * 560 + 255) / 256, 256, 0, stream>>>(
        style, kw_w, kw_b, vw_w, vw_b, aw_w, aw_b, kgb, vgb, agb);

    // 3) instance-norm stats on kv channels (padded stride)
    k_stats<<<Bn * CKV, 256, 0, stream>>>(kv, mu1, rinv1, CKV, CKVP);

    // 4) zero lattice, then splat
    hipMemsetAsync(z, 0, (size_t)Bn * Hh * Ff * S3 * 4, stream);
    k_splat<<<(Bn * Hh * Nn) / 256, 256, 0, stream>>>(
        kv, orig, T_w, T_b, scal, kgb, vgb, mu1, rinv1, idxbuf, wbuf, z);

    // 6) grouped conv: persistent blocks, async double-buffered LDS + WMMA
    k_conv<<<32 * 32, 256, 0, stream>>>(wA, z, conv_b, zc);

    // 7) slice
    k_slice<<<(Bn * Hh * Nn) / 256, 256, 0, stream>>>(idxbuf, wbuf, zc, sliced);

    // 8) stats on sliced
    k_stats<<<Bn * CHF, 256, 0, stream>>>(sliced, mu2, rinv2, CHF, CHF);

    // 9) final AdaIN + ReLU
    k_final<<<(int)(((size_t)Bn * CHF * Nn) / 256), 256, 0, stream>>>(
        sliced, agb, mu2, rinv2, (float*)d_out);

    (void)in_sizes; (void)n_in; (void)out_size; (void)ws_size;
}
